// RecurrentStateSpaceModel_3925600108772
// MI455X (gfx1250) — matrix operand order repaired and checked by reading, not compiled, not run on hardware
//
#include <hip/hip_runtime.h>
#include <hip/hip_bf16.h>
#include <math.h>

// ---------------------------------------------------------------------------
// RSSM on MI455X (gfx1250): bf16 WMMA GEMMs, weights pre-packed into WMMA
// B-fragment order so the inner loop is b128 loads + ds_load_b128 + v_wmma.
// B=256, T=64.
// ---------------------------------------------------------------------------

#define Bsz 256
#define Tsz 64

typedef __attribute__((ext_vector_type(16))) __bf16 v16bf;
typedef __attribute__((ext_vector_type(8)))  float  v8f;

union FragBF { v16bf v; unsigned short u[16]; uint4 q[2]; };

__device__ __forceinline__ unsigned short f32_to_bf16(float f) {
  unsigned int u = __float_as_uint(f);
  unsigned int r = u + 0x7FFFu + ((u >> 16) & 1u);   // round-to-nearest-even
  return (unsigned short)(r >> 16);
}

// ---------------------------------------------------------------------------
// Weight packing: B[K,N] f32 -> bf16 tiles [K/32][N/16][512] where within a
// tile, lane l's 16 fragment elements (col = l&15, k = (l>>4)*16 + e) are the
// 16 consecutive ushorts at offset l*16.  One fragment load = 2x b128.
// ---------------------------------------------------------------------------
__global__ void k_pack_b(const float* __restrict__ src,
                         unsigned short* __restrict__ dst, int K, int N) {
  int idx = blockIdx.x * blockDim.x + threadIdx.x;
  if (idx >= K * N) return;
  const int n16t = N >> 4;
  const int e    = idx & 15;
  const int l    = (idx >> 4) & 31;
  const int tile = idx >> 9;
  const int n16  = tile % n16t;
  const int k32  = tile / n16t;
  const int col  = (n16 << 4) + (l & 15);
  const int k    = (k32 << 5) + ((l >> 4) << 4) + e;
  dst[idx] = f32_to_bf16(src[(size_t)k * N + col]);
}

// ---------------------------------------------------------------------------
// WMMA GEMM: C[M,N] = A[M,K](bf16,row-major) * Bp(packed bf16) + bias.
// Block = 256 threads (8 waves) -> 128x64 C tile; wave w owns rows
// [16w,16w+16) with 4 N-fragments (A fragment reused 4x per K-slice).
// Double-buffered LDS staging of the 4KB packed B slice; 1 barrier / slice.
// MODE: 0 f32 store | 1 bf16 store | 2 tanh->bf16 store |
//       3 obs scatter out[B,T,640] cols[0,512) | 4 smu/exp(sls) scatter
// ---------------------------------------------------------------------------
template<int MODE>
__global__ __launch_bounds__(256) void gemm_bf16_wmma(
    const unsigned short* __restrict__ A, int lda,
    const unsigned short* __restrict__ Bp, int N,
    const float* __restrict__ bias,
    float* __restrict__ Cf, unsigned short* __restrict__ Cb, int ldc,
    int K)
{
  __shared__ __align__(16) unsigned short lds_b[2][2048];   // 2 x 4KB

  const int tid  = threadIdx.x;
  const int lane = tid & 31;
  const int wave = tid >> 5;
  const int half = lane >> 4;
  const int l15  = lane & 15;
  const int ntile = blockIdx.x << 6;
  const int mtile = blockIdx.y << 7;
  const int n16t  = N >> 4;

  v8f acc0 = {}, acc1 = {}, acc2 = {}, acc3 = {};

  const unsigned short* arow =
      A + (size_t)(mtile + (wave << 4) + l15) * lda + (half << 4);

  // uint4 index of this thread's share of the packed B slice for k32 = 0
  const uint4* bq = (const uint4*)Bp;
  size_t btile = (size_t)(ntile >> 4) * 64 + tid;
  const size_t bstep = (size_t)n16t * 64;          // uint4 per k32 row of tiles

  uint4 breg = bq[btile];                          // stage slice 0
  int buf = 0;

  for (int kt = 0; kt < K; kt += 32) {
    *((uint4*)lds_b[buf] + tid) = breg;            // ds_store_b128
    __syncthreads();
    if (kt + 32 < K)                               // stage next slice
      breg = bq[btile + (size_t)((kt >> 5) + 1) * bstep];
    if (kt + 64 < K)                               // global_prefetch_b8
      __builtin_prefetch(bq + btile + (size_t)((kt >> 5) + 2) * bstep, 0, 1);

    // A fragment (16x32): two b128 from L2
    FragBF a;
    a.q[0] = *(const uint4*)(arow + kt);
    a.q[1] = *(const uint4*)(arow + kt + 8);

    // B fragments: 2x ds_load_b128 each, already in fragment order
    const uint4* lq = (const uint4*)lds_b[buf] + (lane << 1);
    FragBF b0, b1, b2, b3;
    b0.q[0] = lq[0];   b0.q[1] = lq[1];
    b1.q[0] = lq[64];  b1.q[1] = lq[65];
    b2.q[0] = lq[128]; b2.q[1] = lq[129];
    b3.q[0] = lq[192]; b3.q[1] = lq[193];

    acc0 = __builtin_amdgcn_wmma_f32_16x16x32_bf16(false, a.v, false, b0.v,
                                                   (short)0, acc0, false, false);
    acc1 = __builtin_amdgcn_wmma_f32_16x16x32_bf16(false, a.v, false, b1.v,
                                                   (short)0, acc1, false, false);
    acc2 = __builtin_amdgcn_wmma_f32_16x16x32_bf16(false, a.v, false, b2.v,
                                                   (short)0, acc2, false, false);
    acc3 = __builtin_amdgcn_wmma_f32_16x16x32_bf16(false, a.v, false, b3.v,
                                                   (short)0, acc3, false, false);
    buf ^= 1;
  }

  v8f accs[4] = {acc0, acc1, acc2, acc3};
#pragma unroll
  for (int nf = 0; nf < 4; ++nf) {
    const int col = ntile + (nf << 4) + l15;
    const float bv = bias ? bias[col] : 0.0f;
#pragma unroll
    for (int v = 0; v < 8; ++v) {
      const int row = mtile + (wave << 4) + (half << 3) + v;
      float val = accs[nf][v] + bv;
      if (MODE == 0) {
        Cf[(size_t)row * ldc + col] = val;
      } else if (MODE == 1) {
        Cb[(size_t)row * ldc + col] = f32_to_bf16(val);
      } else if (MODE == 2) {
        Cb[(size_t)row * ldc + col] = f32_to_bf16(tanhf(val));
      } else if (MODE == 3) {
        const int bb = row & (Bsz - 1);            // din rows are t*B + b
        const int tt = row >> 8;
        Cf[(size_t)(bb * Tsz + tt) * 640 + col] = val;
      } else if (MODE == 4) {
        const int bb = row & (Bsz - 1);
        const int tt = row >> 8;
        const size_t base = (size_t)(bb * Tsz + tt) * 640;
        if (col < 64) Cf[base + 512 + col] = val;
        else          Cf[base + 576 + (col - 64)] = expf(val);
      }
    }
  }
}

// ---------------------------------------------------------------------------
// Elementwise helpers
// ---------------------------------------------------------------------------
__global__ void k_f32_to_bf16(const float* __restrict__ s,
                              unsigned short* __restrict__ d, int n) {
  int i = blockIdx.x * blockDim.x + threadIdx.x;
  if (i < n) d[i] = f32_to_bf16(s[i]);
}

__global__ void k_zero_u32(unsigned int* __restrict__ p, int n) {
  int i = blockIdx.x * blockDim.x + threadIdx.x;
  if (i < n) p[i] = 0u;
}

// act MLP layer 1 (K=6, too small for WMMA): ae_h = tanh(a_t @ w1 + b1), bf16
__global__ void k_amlp1(const float* __restrict__ act, int t,
                        const float* __restrict__ w1,
                        const float* __restrict__ b1,
                        unsigned short* __restrict__ aeh) {
  int j = blockIdx.x * blockDim.x + threadIdx.x;   // 256*512
  int b = j >> 9, o = j & 511;
  const float* ap = act + ((size_t)b * Tsz + t) * 6;
  float s = b1[o];
#pragma unroll
  for (int i = 0; i < 6; ++i) s = fmaf(ap[i], w1[i * 512 + o], s);
  aeh[j] = f32_to_bf16(tanhf(s));
}

// posterior input (h0=0 rows dropped): [state(64) | enc_mu(128)] -> bf16
__global__ void k_build_pin(const float* __restrict__ state,
                            const unsigned short* __restrict__ encout,
                            unsigned short* __restrict__ pin) {
  int j = blockIdx.x * blockDim.x + threadIdx.x;   // 256*192
  int b = j / 192, c = j % 192;
  pin[j] = (c < 64) ? f32_to_bf16(state[b * 64 + c])
                    : encout[b * 256 + (c - 64)];  // mu = first 128 of 256
}

// reparameterized sample: z = mu + exp(ls)*eps; bf16 into x (GRU input) and
// optionally into din z-region (decoder history)
__global__ void k_sample(const float* __restrict__ lout,
                         const float* __restrict__ eps,
                         unsigned short* __restrict__ xz,
                         unsigned short* __restrict__ dinz) {
  int j = blockIdx.x * blockDim.x + threadIdx.x;   // 256*1024
  int b = j >> 10, l = j & 1023;
  float mu = lout[(size_t)b * 2048 + l];
  float ls = lout[(size_t)b * 2048 + 1024 + l];
  float z = fmaf(expf(ls), eps[j], mu);
  unsigned short zb = f32_to_bf16(z);
  xz[(size_t)b * 1536 + l] = zb;
  if (dinz) dinz[(size_t)b * 2048 + l] = zb;
}

// torch GRUCell gate math; h kept in f32 for the recurrence, bf16 copy in din
__global__ void k_gate(const float* __restrict__ gi,
                       const float* __restrict__ gh,
                       float* __restrict__ hf,
                       unsigned short* __restrict__ dinh) {
  int j = blockIdx.x * blockDim.x + threadIdx.x;   // 256*1024
  int b = j >> 10, l = j & 1023;
  const float* gib = gi + (size_t)b * 3072;
  const float* ghb = gh + (size_t)b * 3072;
  float r = 1.0f / (1.0f + expf(-(gib[l]        + ghb[l])));
  float u = 1.0f / (1.0f + expf(-(gib[1024 + l] + ghb[1024 + l])));
  float n = tanhf(gib[2048 + l] + r * ghb[2048 + l]);
  float h = (1.0f - u) * n + u * hf[j];
  hf[j] = h;
  dinh[(size_t)b * 2048 + l] = f32_to_bf16(h);
}

// ---------------------------------------------------------------------------
extern "C" void kernel_launch(void* const* d_in, const int* in_sizes, int n_in,
                              void* d_out, int out_size, void* d_ws,
                              size_t ws_size, hipStream_t stream) {
  const float* obs_feat = (const float*)d_in[0];
  const float* state    = (const float*)d_in[1];
  const float* act      = (const float*)d_in[2];
  const float* eps0     = (const float*)d_in[3];
  const float* eps_seq  = (const float*)d_in[4];
  const float* enc_w1 = (const float*)d_in[5];  const float* enc_b1 = (const float*)d_in[6];
  const float* enc_w2 = (const float*)d_in[7];  const float* enc_b2 = (const float*)d_in[8];
  const float* post_w1 = (const float*)d_in[9]; const float* post_b1 = (const float*)d_in[10];
  const float* post_w2 = (const float*)d_in[11];const float* post_b2 = (const float*)d_in[12];
  const float* amlp_w1 = (const float*)d_in[13];const float* amlp_b1 = (const float*)d_in[14];
  const float* amlp_w2 = (const float*)d_in[15];const float* amlp_b2 = (const float*)d_in[16];
  const float* gru_w_ih = (const float*)d_in[17];
  const float* gru_w_hh = (const float*)d_in[18];
  const float* gru_b_ih = (const float*)d_in[19];
  const float* gru_b_hh = (const float*)d_in[20];
  const float* lsd_w1 = (const float*)d_in[21]; const float* lsd_b1 = (const float*)d_in[22];
  const float* lsd_w2 = (const float*)d_in[23]; const float* lsd_b2 = (const float*)d_in[24];
  const float* dec_w1 = (const float*)d_in[25]; const float* dec_b1 = (const float*)d_in[26];
  const float* dec_w2 = (const float*)d_in[27]; const float* dec_b2 = (const float*)d_in[28];
  const float* sd_w = (const float*)d_in[29];   const float* sd_b = (const float*)d_in[30];
  float* out = (float*)d_out;

  char* ws = (char*)d_ws;
  size_t off = 0;
  auto alloc = [&](size_t bytes) -> void* {
    void* p = ws + off;
    off += (bytes + 255) & ~(size_t)255;
    return p;
  };

  // packed bf16 weights (fit in L2 -> recurrent loop streams from L2)
  unsigned short* w_enc1 = (unsigned short*)alloc((size_t)512 * 512 * 2);
  unsigned short* w_enc2 = (unsigned short*)alloc((size_t)512 * 256 * 2);
  unsigned short* w_post1 = (unsigned short*)alloc((size_t)192 * 1024 * 2);
  unsigned short* w_post2 = (unsigned short*)alloc((size_t)1024 * 2048 * 2);
  unsigned short* w_amlp2 = (unsigned short*)alloc((size_t)512 * 512 * 2);
  unsigned short* w_gih   = (unsigned short*)alloc((size_t)1536 * 3072 * 2);
  unsigned short* w_ghh   = (unsigned short*)alloc((size_t)1024 * 3072 * 2);
  unsigned short* w_lsd1  = (unsigned short*)alloc((size_t)1024 * 1024 * 2);
  unsigned short* w_lsd2  = (unsigned short*)alloc((size_t)1024 * 2048 * 2);
  unsigned short* w_dec1  = (unsigned short*)alloc((size_t)2048 * 1024 * 2);
  unsigned short* w_dec2  = (unsigned short*)alloc((size_t)1024 * 512 * 2);
  unsigned short* w_sd    = (unsigned short*)alloc((size_t)2048 * 128 * 2);
  // activations
  unsigned short* obs_bf = (unsigned short*)alloc((size_t)Bsz * 512 * 2);
  unsigned short* ench   = (unsigned short*)alloc((size_t)Bsz * 512 * 2);
  unsigned short* encout = (unsigned short*)alloc((size_t)Bsz * 256 * 2);
  unsigned short* pin    = (unsigned short*)alloc((size_t)Bsz * 192 * 2);
  unsigned short* posth  = (unsigned short*)alloc((size_t)Bsz * 1024 * 2);
  float*          lout   = (float*)alloc((size_t)Bsz * 2048 * 4);   // pout/lout
  unsigned short* x      = (unsigned short*)alloc((size_t)Bsz * 1536 * 2);
  unsigned short* aeh    = (unsigned short*)alloc((size_t)Bsz * 512 * 2);
  float*          gi     = (float*)alloc((size_t)Bsz * 3072 * 4);
  float*          gh     = (float*)alloc((size_t)Bsz * 3072 * 4);
  float*          hf     = (float*)alloc((size_t)Bsz * 1024 * 4);
  unsigned short* zbf    = (unsigned short*)alloc((size_t)Bsz * 1024 * 2); // h0=0
  unsigned short* lsdh   = (unsigned short*)alloc((size_t)Bsz * 1024 * 2);
  unsigned short* din    = (unsigned short*)alloc((size_t)Tsz * Bsz * 2048 * 2);
  unsigned short* dh     = (unsigned short*)alloc((size_t)Tsz * Bsz * 1024 * 2);
  (void)ws_size; (void)in_sizes; (void)n_in; (void)out_size;

  auto pack = [&](const float* s, unsigned short* d, int K, int N) {
    k_pack_b<<<(K * N + 255) / 256, 256, 0, stream>>>(s, d, K, N);
  };
  auto gemm = [&](int mode, const unsigned short* A, int lda,
                  const unsigned short* Bp, const float* bias,
                  void* C, int ldc, int M, int N, int K) {
    dim3 g(N / 64, M / 128), blk(256);
    float* Cf = (float*)C;
    unsigned short* Cb = (unsigned short*)C;
    switch (mode) {
      case 0: gemm_bf16_wmma<0><<<g, blk, 0, stream>>>(A, lda, Bp, N, bias, Cf, Cb, ldc, K); break;
      case 1: gemm_bf16_wmma<1><<<g, blk, 0, stream>>>(A, lda, Bp, N, bias, Cf, Cb, ldc, K); break;
      case 2: gemm_bf16_wmma<2><<<g, blk, 0, stream>>>(A, lda, Bp, N, bias, Cf, Cb, ldc, K); break;
      case 3: gemm_bf16_wmma<3><<<g, blk, 0, stream>>>(A, lda, Bp, N, bias, Cf, Cb, ldc, K); break;
      case 4: gemm_bf16_wmma<4><<<g, blk, 0, stream>>>(A, lda, Bp, N, bias, Cf, Cb, ldc, K); break;
    }
  };

  // ---- weight packing / input conversion ------------------------------
  pack(enc_w1, w_enc1, 512, 512);
  pack(enc_w2, w_enc2, 512, 256);
  pack(post_w1 + (size_t)1024 * 1024, w_post1, 192, 1024);  // h0=0 rows dropped
  pack(post_w2, w_post2, 1024, 2048);
  pack(amlp_w2, w_amlp2, 512, 512);
  pack(gru_w_ih, w_gih, 1536, 3072);
  pack(gru_w_hh, w_ghh, 1024, 3072);
  pack(lsd_w1, w_lsd1, 1024, 1024);
  pack(lsd_w2, w_lsd2, 1024, 2048);
  pack(dec_w1, w_dec1, 2048, 1024);
  pack(dec_w2, w_dec2, 1024, 512);
  pack(sd_w, w_sd, 2048, 128);
  k_f32_to_bf16<<<(Bsz * 512 + 255) / 256, 256, 0, stream>>>(obs_feat, obs_bf, Bsz * 512);
  k_zero_u32<<<(Bsz * 1024 / 2 + 255) / 256, 256, 0, stream>>>((unsigned int*)zbf, Bsz * 1024 / 2);
  k_zero_u32<<<(Bsz * 1024 + 255) / 256, 256, 0, stream>>>((unsigned int*)hf, Bsz * 1024);

  // ---- encoder + posterior (h0=0 -> only last 192 input cols matter) ---
  gemm(2, obs_bf, 512, w_enc1, enc_b1, ench, 512, Bsz, 512, 512);
  gemm(1, ench, 512, w_enc2, enc_b2, encout, 256, Bsz, 256, 512);
  k_build_pin<<<(Bsz * 192) / 256, 256, 0, stream>>>(state, encout, pin);
  gemm(2, pin, 192, w_post1, post_b1, posth, 1024, Bsz, 1024, 192);
  gemm(0, posth, 1024, w_post2, post_b2, lout, 2048, Bsz, 2048, 1024);
  k_sample<<<1024, 256, 0, stream>>>(lout, eps0, x, (unsigned short*)nullptr);

  // ---- sequential GRU recurrence --------------------------------------
  for (int t = 0; t < Tsz; ++t) {
    unsigned short* dint = din + (size_t)t * Bsz * 2048;
    k_amlp1<<<512, 256, 0, stream>>>(act, t, amlp_w1, amlp_b1, aeh);
    // ae -> x cols [1024,1536)
    gemm(1, aeh, 512, w_amlp2, amlp_b2, x + 1024, 1536, Bsz, 512, 512);
    // gi = [z|ae] @ W_ih + b_ih
    gemm(0, x, 1536, w_gih, gru_b_ih, gi, 3072, Bsz, 3072, 1536);
    // gh = h_prev @ W_hh + b_hh (h_prev lives in din[t-1] h-region; t=0 -> 0)
    const unsigned short* hprev = (t == 0) ? zbf : (din + (size_t)(t - 1) * Bsz * 2048 + 1024);
    const int hlda = (t == 0) ? 1024 : 2048;
    gemm(0, hprev, hlda, w_ghh, gru_b_hh, gh, 3072, Bsz, 3072, 1024);
    k_gate<<<1024, 256, 0, stream>>>(gi, gh, hf, dint + 1024);
    // latent state decoder on h_new
    gemm(2, dint + 1024, 2048, w_lsd1, lsd_b1, lsdh, 1024, Bsz, 1024, 1024);
    gemm(0, lsdh, 1024, w_lsd2, lsd_b2, lout, 2048, Bsz, 2048, 1024);
    // z_{t+1} -> x z-region (next step) and din[t] z-region (history)
    k_sample<<<1024, 256, 0, stream>>>(lout, eps_seq + (size_t)t * Bsz * 1024, x, dint);
  }

  // ---- decoders over all B*T rows, scattered straight into out[B,T,640]
  gemm(2, din, 2048, w_dec1, dec_b1, dh, 1024, Tsz * Bsz, 1024, 2048);
  gemm(3, dh, 1024, w_dec2, dec_b2, out, 0, Tsz * Bsz, 512, 1024);
  gemm(4, din, 2048, w_sd, sd_b, out, 0, Tsz * Bsz, 128, 2048);
}
